// Attention_89867895701940
// MI455X (gfx1250) — matrix operand order repaired and checked by reading, not compiled, not run on hardware
//
#include <hip/hip_runtime.h>
#include <hip/hip_bf16.h>

// ---------- types ----------
typedef __attribute__((ext_vector_type(16))) __bf16 v16bf;
typedef __attribute__((ext_vector_type(2)))  __bf16 v2bf;
typedef __attribute__((ext_vector_type(8)))  float  v8f;
typedef __attribute__((ext_vector_type(4)))  unsigned int u32x4;
typedef __attribute__((ext_vector_type(4)))  float  f32x4;

union Frag { u32x4 q[2]; v16bf v; };   // 32B: one WMMA A or B fragment (16 bf16)

// f32 -> bf16 (RNE) via native conversion hardware; packed pair in one u32
__device__ __forceinline__ unsigned int pack_bf16x2(float lo, float hi) {
  v2bf p;
  p.x = (__bf16)lo;
  p.y = (__bf16)hi;
  return __builtin_bit_cast(unsigned int, p);
}
__device__ __forceinline__ unsigned short f2bf(float f) {
  __bf16 h = (__bf16)f;
  return __builtin_bit_cast(unsigned short, h);
}

#define NROWS  1536         // B*C = 4*384
#define NN     16384        // H*W
#define HD     48
#define SPLITK 16           // K chunks for the attention GEMM (1024 K each)

// ---------- kernel 1: row L2 norms (inv = 1/max(||row||,eps)) ----------
__launch_bounds__(256)
__global__ void norm_kernel(const float* __restrict__ in1, const float* __restrict__ in2,
                            float* __restrict__ inv_q, float* __restrict__ inv_k) {
  const int  row = blockIdx.x & (NROWS - 1);
  const bool isK = blockIdx.x >= NROWS;
  const float* src = (isK ? in2 : in1) + (size_t)row * NN;
  float ss = 0.f;
  for (int i = threadIdx.x; i < NN / 4; i += 256) {
    f32x4 v = ((const f32x4*)src)[i];
    ss += v.x * v.x + v.y * v.y + v.z * v.z + v.w * v.w;
  }
  __shared__ float red[256];
  red[threadIdx.x] = ss;
  __syncthreads();
  for (int s = 128; s > 0; s >>= 1) {
    if (threadIdx.x < s) red[threadIdx.x] += red[threadIdx.x + s];
    __syncthreads();
  }
  if (threadIdx.x == 0) {
    float inv = 1.0f / fmaxf(sqrtf(red[0]), 1e-12f);
    (isK ? inv_k : inv_q)[row] = inv;
  }
}

// ---------- kernel 2: split-K partial S = qn @ kn^T  (per b,h: 48x48, K=16384) ----------
// grid: 32 (b*h) * SPLITK blocks, 288 threads (9 waves, one 16x16 tile each)
// Double-buffered LDS: load chunk i+1 into regs while computing chunk i.
__launch_bounds__(288)
__global__ void attn_gemm_kernel(const float* __restrict__ in1, const float* __restrict__ in2,
                                 const float* __restrict__ inv_q, const float* __restrict__ inv_k,
                                 float* __restrict__ Spart) {
  const int bh    = blockIdx.x >> 4;            // 0..31
  const int chunk = blockIdx.x & (SPLITK - 1);  // 0..15 (1024 K each)
  const int tid   = threadIdx.x;
  const int lane  = tid & 31;
  const int wave  = tid >> 5;                   // 0..8
  const int wm = wave / 3, wn = wave % 3;

  __shared__ unsigned int Qs[2][48][64];  // 2 x (48 x 128 bf16, packed pairs)
  __shared__ unsigned int Ks[2][48][64];

  const size_t rowBase = (size_t)bh * HD;
  const float* qb = in1 + rowBase * NN;
  const float* kb = in2 + rowBase * NN;
  const int k0c = chunk * (NN / SPLITK);

  f32x4 qR[6], kR[6];
  auto load_regs = [&](int kk) {
#pragma unroll
    for (int l = 0; l < 6; ++l) {
      int idx = tid + l * 288;
      if (l < 5 || tid < 96) {                 // 1536 float4 total
        int r = idx >> 5, c4 = idx & 31;
        size_t off = (size_t)r * NN + k0c + kk + c4 * 4;
        qR[l] = *(const f32x4*)(qb + off);
        kR[l] = *(const f32x4*)(kb + off);
      }
    }
  };
  auto store_lds = [&](int buf) {
#pragma unroll
    for (int l = 0; l < 6; ++l) {
      int idx = tid + l * 288;
      if (l < 5 || tid < 96) {
        int r = idx >> 5, c4 = idx & 31;
        float iq = inv_q[rowBase + r];
        float ik = inv_k[rowBase + r];
        Qs[buf][r][c4 * 2]     = pack_bf16x2(qR[l].x * iq, qR[l].y * iq);
        Qs[buf][r][c4 * 2 + 1] = pack_bf16x2(qR[l].z * iq, qR[l].w * iq);
        Ks[buf][r][c4 * 2]     = pack_bf16x2(kR[l].x * ik, kR[l].y * ik);
        Ks[buf][r][c4 * 2 + 1] = pack_bf16x2(kR[l].z * ik, kR[l].w * ik);
      }
    }
  };

  const int am   = wm * 16 + (lane & 15);
  const int bn   = wn * 16 + (lane & 15);
  const int aoff = (lane < 16) ? 0 : 4;   // A: K 0..7 / 8..15 (uint pairs)
  const int boff = (lane < 16) ? 0 : 4;   // B: K 0..7 / 8..15 (uint pairs)

  v8f acc = {};
  load_regs(0);
  store_lds(0);
  __syncthreads();
  const int NITER = (NN / SPLITK) / 128;   // 8
  for (int it = 0; it < NITER; ++it) {
    const int  cur  = it & 1;
    const bool more = (it + 1) < NITER;
    if (more) load_regs((it + 1) * 128);   // global loads overlap WMMA below
#pragma unroll
    for (int s = 0; s < 4; ++s) {
      Frag A, Bf;
      A.q[0]  = *(const u32x4*)&Qs[cur][am][s * 16 + aoff];
      A.q[1]  = *(const u32x4*)&Qs[cur][am][s * 16 + aoff + 8];
      Bf.q[0] = *(const u32x4*)&Ks[cur][bn][s * 16 + boff];
      Bf.q[1] = *(const u32x4*)&Ks[cur][bn][s * 16 + boff + 8];
      acc = __builtin_amdgcn_wmma_f32_16x16x32_bf16(false, A.v, false, Bf.v,
                                                    (short)0, acc, false, false);
    }
    if (more) store_lds(cur ^ 1);          // writes other buffer: no race with reads
    __syncthreads();
  }

  float* Sp = Spart + ((size_t)chunk * 32 + bh) * (HD * HD);
  const int col   = wn * 16 + (lane & 15);
  const int rbase = wm * 16 + ((lane < 16) ? 0 : 8);
#pragma unroll
  for (int r = 0; r < 8; ++r) Sp[(rbase + r) * HD + col] = acc[r];
}

// ---------- kernel 3: reduce split-K + temperature + softmax -> attn ----------
__launch_bounds__(256)
__global__ void softmax_kernel(const float* __restrict__ Spart,
                               const float* __restrict__ temperature,
                               float* __restrict__ attn) {
  const int bh = blockIdx.x;       // 0..31
  const int h  = bh & 7;
  __shared__ float Ss[HD * HD];
  for (int idx = threadIdx.x; idx < HD * HD; idx += 256) {
    float s = 0.f;
#pragma unroll
    for (int c = 0; c < SPLITK; ++c) s += Spart[((size_t)c * 32 + bh) * (HD * HD) + idx];
    Ss[idx] = s;
  }
  __syncthreads();
  if (threadIdx.x < HD) {
    const float t = temperature[h];
    const float* row = &Ss[threadIdx.x * HD];
    float m = -3.4e38f;
    for (int d = 0; d < HD; ++d) m = fmaxf(m, row[d] * t);
    float sum = 0.f;
    for (int d = 0; d < HD; ++d) sum += __expf(row[d] * t - m);
    float inv = 1.0f / sum;
    float* out = attn + (size_t)bh * (HD * HD) + threadIdx.x * HD;
    for (int d = 0; d < HD; ++d) out[d] = __expf(row[d] * t - m) * inv;
  }
}

// ---------- kernel 4: fold projection into attention: W2[b] = proj_w ⊛ attn[b] ----------
// W2[b][o][h*48+e] = sum_d proj_w[o][h*48+d] * attn[b][h][d][e]
__launch_bounds__(384)
__global__ void fold_kernel(const float* __restrict__ attn, const float* __restrict__ proj_w,
                            float* __restrict__ W2) {
  const int bh = blockIdx.x;            // 0..31
  const int b = bh >> 3, h = bh & 7;
  const int o = threadIdx.x;            // 0..383
  __shared__ float As[HD * HD];
  for (int idx = threadIdx.x; idx < HD * HD; idx += 384)
    As[idx] = attn[(size_t)bh * (HD * HD) + idx];
  __syncthreads();
  float pw[HD];
  const float* wrow = proj_w + (size_t)o * 384 + h * HD;
#pragma unroll
  for (int d = 0; d < HD; ++d) pw[d] = wrow[d];
  float* out = W2 + ((size_t)b * 384 + o) * 384 + h * HD;
  for (int e = 0; e < HD; ++e) {
    float s = 0.f;
#pragma unroll
    for (int d = 0; d < HD; ++d) s += pw[d] * As[d * HD + e];
    out[e] = s;
  }
}

// ---------- kernel 5: final = W2[b] (384x384) @ kn[b] (384x16384) + bias ----------
// grid: 4 * 3 * 128 blocks (128x128 output tiles), 256 threads, 8 waves x (16x128)
// Double-buffered LDS ping-pong over the 12 K-chunks.
__launch_bounds__(256)
__global__ void out_gemm_kernel(const float* __restrict__ in2, const float* __restrict__ inv_k,
                                const float* __restrict__ W2, const float* __restrict__ proj_b,
                                float* __restrict__ out) {
  const int bid   = blockIdx.x;
  const int batch = bid / 384;
  const int rem   = bid % 384;
  const int m0 = (rem / 128) * 128;
  const int n0 = (rem % 128) * 128;
  const int tid  = threadIdx.x;
  const int lane = tid & 31;
  const int wave = tid >> 5;           // 0..7 -> rows m0+wave*16..+16

  __shared__ unsigned int   Asm[2][128][16];   // W2 tile: 128 x 32 bf16 (row-major packed)
  __shared__ unsigned short BtS[2][128][32];   // kn tile TRANSPOSED: [n][c]

  const float* Ab  = W2  + (size_t)batch * 384 * 384;
  const float* Bb  = in2 + (size_t)batch * 384 * NN;
  const float* ivk = inv_k + batch * 384;

  f32x4 aR[4], bR[4];
  auto load_regs = [&](int k0) {
#pragma unroll
    for (int i = 0; i < 4; ++i) {
      int idx = tid + i * 256;
      int r = idx >> 3, c4 = idx & 7;              // A: 128 rows x 8 float4
      aR[i] = *(const f32x4*)(Ab + (size_t)(m0 + r) * 384 + k0 + c4 * 4);
      int c = idx >> 5, n4 = idx & 31;             // B: 32 rows x 32 float4
      bR[i] = *(const f32x4*)(Bb + (size_t)(k0 + c) * NN + n0 + n4 * 4);
    }
  };
  auto store_lds = [&](int buf, int k0) {
#pragma unroll
    for (int i = 0; i < 4; ++i) {
      int idx = tid + i * 256;
      int r = idx >> 3, c4 = idx & 7;
      Asm[buf][r][c4 * 2]     = pack_bf16x2(aR[i].x, aR[i].y);
      Asm[buf][r][c4 * 2 + 1] = pack_bf16x2(aR[i].z, aR[i].w);
      int c = idx >> 5, n4 = idx & 31;
      float ik = ivk[k0 + c];
      BtS[buf][n4 * 4 + 0][c] = f2bf(bR[i].x * ik);
      BtS[buf][n4 * 4 + 1][c] = f2bf(bR[i].y * ik);
      BtS[buf][n4 * 4 + 2][c] = f2bf(bR[i].z * ik);
      BtS[buf][n4 * 4 + 3][c] = f2bf(bR[i].w * ik);
    }
  };

  // init accumulators with bias (row-dependent only)
  v8f acc[8];
  {
    const int rb = (lane < 16) ? 0 : 8;
#pragma unroll
    for (int r = 0; r < 8; ++r) {
      float bv = proj_b[m0 + wave * 16 + rb + r];
#pragma unroll
      for (int nf = 0; nf < 8; ++nf) acc[nf][r] = bv;
    }
  }

  const int am   = wave * 16 + (lane & 15);
  const int aoff = (lane < 16) ? 0 : 4;
  const int kh   = (lane < 16) ? 0 : 8;

  load_regs(0);
  store_lds(0, 0);
  __syncthreads();
  for (int kc = 0; kc < 12; ++kc) {              // 384 / 32
    const int  cur  = kc & 1;
    const bool more = (kc + 1) < 12;
    if (more) load_regs((kc + 1) * 32);          // overlap global latency with WMMA
    Frag A;
    A.q[0] = *(const u32x4*)&Asm[cur][am][aoff];
    A.q[1] = *(const u32x4*)&Asm[cur][am][aoff + 8];
#pragma unroll
    for (int nf = 0; nf < 8; ++nf) {
      const int n = nf * 16 + (lane & 15);
      Frag Bf;
      Bf.q[0] = *(const u32x4*)&BtS[cur][n][kh];
      Bf.q[1] = *(const u32x4*)&BtS[cur][n][kh + 16];
      acc[nf] = __builtin_amdgcn_wmma_f32_16x16x32_bf16(false, A.v, false, Bf.v,
                                                        (short)0, acc[nf], false, false);
    }
    if (more) store_lds(cur ^ 1, (kc + 1) * 32); // other buffer: no race
    __syncthreads();
  }

  float* ob = out + ((size_t)batch * 384 + m0 + wave * 16) * NN + n0;
  const int rb = (lane < 16) ? 0 : 8;
  const int cl = lane & 15;
#pragma unroll
  for (int nf = 0; nf < 8; ++nf)
#pragma unroll
    for (int r = 0; r < 8; ++r)
      ob[(size_t)(rb + r) * NN + nf * 16 + cl] = acc[nf][r];
}

// ---------- launch ----------
extern "C" void kernel_launch(void* const* d_in, const int* in_sizes, int n_in,
                              void* d_out, int out_size, void* d_ws, size_t ws_size,
                              hipStream_t stream) {
  (void)in_sizes; (void)n_in; (void)out_size; (void)ws_size;
  const float* in1         = (const float*)d_in[0];
  const float* in2         = (const float*)d_in[1];
  const float* temperature = (const float*)d_in[2];
  const float* proj_w      = (const float*)d_in[3];
  const float* proj_b      = (const float*)d_in[4];
  float* out = (float*)d_out;

  // workspace layout (floats): inv_q | inv_k | attn | (Spart aliased with W2)
  float* ws    = (float*)d_ws;
  float* inv_q = ws;                       // 1536
  float* inv_k = ws + 1536;                // 1536
  float* attn  = ws + 3072;                // 32*48*48 = 73728
  float* Spart = ws + 3072 + 73728;        // SPLITK*32*48*48 = 1179648
  float* W2    = Spart;                    // aliased: lifetimes don't overlap

  norm_kernel<<<2 * NROWS, 256, 0, stream>>>(in1, in2, inv_q, inv_k);
  attn_gemm_kernel<<<32 * SPLITK, 288, 0, stream>>>(in1, in2, inv_q, inv_k, Spart);
  softmax_kernel<<<32, 256, 0, stream>>>(Spart, temperature, attn);
  fold_kernel<<<32, 384, 0, stream>>>(attn, proj_w, W2);
  out_gemm_kernel<<<4 * 3 * 128, 256, 0, stream>>>(in2, inv_k, W2, proj_b, out);
}
